// CNF_32985348833502
// MI455X (gfx1250) — matrix operand order repaired and checked by reading, not compiled, not run on hardware
//
#include <hip/hip_runtime.h>

typedef __attribute__((ext_vector_type(16))) __bf16 bf16x16;
typedef __attribute__((ext_vector_type(8)))  __bf16 bf16x8;
typedef __attribute__((ext_vector_type(8)))  float  f32x8;

#define TDIM   64      // theta dim (D)
#define CDIM   128     // context dim
#define HID    512     // hidden
#define INPAD  224     // 193 padded to multiple of 32
#define NEPS   4
#define MROWS  16      // batch rows per workgroup (grid = 4)
#define NSTEP  64      // fixed RK4 steps, t: -1 -> 0

// workspace layout (bf16 element offsets)
#define OFF_W1T   0         // [512][224]  B for L1: W1t[n][k] = W1[k][n] (k>=193 -> 0)
#define OFF_W2T   114688    // [512][512]  B for L2: W2t[n][k] = W2[k][n]
#define OFF_W3T   376832    // [64][512]   B for L3: W3t[n][k] = W3[k][n]
#define OFF_W3B   409600    // [512][64]   B for bwd1 (= bf16(W3) row-major)
#define OFF_W2B   442368    // [512][512]  B for bwd2 (= bf16(W2) row-major)
#define OFF_W1ZB  704512    // [64][512]   B for bwd3 (= bf16(W1[0:64,:]) row-major)
#define OFF_EB    737280    // [4][64][64] eps reordered: Eb[e][b][d]
#define OFF_END   753664

static __device__ __forceinline__ unsigned short f2bf(float f) {
  return __builtin_bit_cast(unsigned short, (__bf16)f);   // hw f32->bf16 cvt
}
static __device__ __forceinline__ float bf2f(unsigned short h) {
  return (float)__builtin_bit_cast(__bf16, h);
}
// gfx1250 hardware tanh (V_TANH_F32)
static __device__ __forceinline__ float htanh(float x) {
#if __has_builtin(__builtin_amdgcn_tanhf)
  return __builtin_amdgcn_tanhf(x);
#elif __has_builtin(__builtin_amdgcn_tanh_f32)
  return __builtin_amdgcn_tanh_f32(x);
#else
  float r;
  // TRANS op: one VALU slot before result use (ISA 7.x hazard rule)
  asm("v_tanh_f32 %0, %1\n\tv_nop" : "=v"(r) : "v"(x));
  return r;
#endif
}
static __device__ __forceinline__ f32x8 wmma_bf16(bf16x16 a, bf16x16 b, f32x8 c) {
  return __builtin_amdgcn_wmma_f32_16x16x32_bf16(false, a, false, b, (short)0, c,
                                                 false, false);
}
// A fragment (16x32 bf16 tile), row-major source with row stride `stride` elems.
static __device__ __forceinline__ bf16x16 load_a16(const unsigned short* base,
                                                   int stride, int k0, int lane) {
  int m = lane & 15, half = lane >> 4;
  const unsigned short* p = base + m * stride + k0 + half * 16;
  bf16x8 lo = *(const bf16x8*)p;
  bf16x8 hi = *(const bf16x8*)(p + 8);
  bf16x16 r;
#pragma unroll
  for (int i = 0; i < 8; ++i) { r[i] = lo[i]; r[8 + i] = hi[i]; }
  return r;
}
// B fragment (32x16 bf16 tile) from transposed [N][K] storage: one 32B load/lane.
static __device__ __forceinline__ bf16x16 load_b16(const unsigned short* bt,
                                                   int stride, int n0, int k0, int lane) {
  int n = n0 + (lane & 15);
  int kb = k0 + (lane >> 4) * 16;
  return *(const bf16x16*)(bt + n * stride + kb);
}

__global__ void cnf_prep(const float* __restrict__ W1, const float* __restrict__ W2,
                         const float* __restrict__ W3, const float* __restrict__ eps,
                         unsigned short* __restrict__ ws) {
  for (long long idx = (long long)blockIdx.x * blockDim.x + threadIdx.x; idx < OFF_END;
       idx += (long long)gridDim.x * blockDim.x) {
    float v;
    if (idx < OFF_W2T) {                 // W1t [512][224]
      long long j = idx;           int n = (int)(j / INPAD), k = (int)(j % INPAD);
      v = (k < 193) ? W1[k * HID + n] : 0.0f;
    } else if (idx < OFF_W3T) {          // W2t [512][512]
      long long j = idx - OFF_W2T; int n = (int)(j / HID), k = (int)(j % HID);
      v = W2[k * HID + n];
    } else if (idx < OFF_W3B) {          // W3t [64][512]
      long long j = idx - OFF_W3T; int n = (int)(j / HID), k = (int)(j % HID);
      v = W3[k * TDIM + n];
    } else if (idx < OFF_W2B) {          // W3b = bf16(W3) flat
      v = W3[idx - OFF_W3B];
    } else if (idx < OFF_W1ZB) {         // W2b = bf16(W2) flat
      v = W2[idx - OFF_W2B];
    } else if (idx < OFF_EB) {           // W1zb = bf16(W1[0:64,:]) flat
      v = W1[idx - OFF_W1ZB];
    } else {                             // Eb[e][b][d] <- eps[b][d][e]
      long long j = idx - OFF_EB;
      int e = (int)(j >> 12), b = (int)((j >> 6) & 63), d = (int)(j & 63);
      v = eps[b * (TDIM * NEPS) + d * NEPS + e];
    }
    ws[idx] = f2bf(v);
  }
}

__global__ void __launch_bounds__(512, 1)
cnf_solve(const float* __restrict__ theta, const float* __restrict__ ctx,
          const float* __restrict__ b1, const float* __restrict__ b2,
          const float* __restrict__ b3, unsigned short* __restrict__ ws,
          float* __restrict__ out) {
  __shared__ __align__(16) unsigned short sA[MROWS * HID];   // X (fwd) / G2' (bwd)
  __shared__ __align__(16) unsigned short sB[MROWS * HID];   // G1' (bwd)
  __shared__ __align__(16) unsigned short sH1[MROWS * HID];  // tanh layer-1 act
  __shared__ __align__(16) unsigned short sH2[MROWS * HID];  // tanh layer-2 act
  __shared__ __align__(16) unsigned short sE[NEPS * MROWS * TDIM]; // eps tiles
  __shared__ float sF[MROWS * TDIM];           // vf output (f32)
  __shared__ float sTR[MROWS];                 // Hutchinson accumulator
  __shared__ float sY[MROWS * TDIM];           // RK4 state z
  __shared__ float sYlp[MROWS];                // RK4 state logp
  __shared__ float sKa[MROWS * TDIM];          // RK4 weighted-k accumulator
  __shared__ float sKalp[MROWS];
  __shared__ float sKc[MROWS * TDIM];          // last stage derivative
  __shared__ float sKclp[MROWS];

  const int tid = threadIdx.x;
  const int lane = tid & 31;
  const int wid = tid >> 5;                    // 16 waves
  const int m0 = blockIdx.x * MROWS;
  const int nlo = lane & 15;
  const int mb = (lane >> 4) * 8;

  const unsigned short* w1t  = ws + OFF_W1T;
  const unsigned short* w2t  = ws + OFF_W2T;
  const unsigned short* w3t  = ws + OFF_W3T;
  const unsigned short* w3b  = ws + OFF_W3B;
  const unsigned short* w2b  = ws + OFF_W2B;
  const unsigned short* w1zb = ws + OFF_W1ZB;

  // ---- stage this WG's eps tiles into LDS with gfx1250 async DMA ----
  // Eb global: [e][64][64] bf16; per WG: 4 chunks of 16x64 (2KB). 512 threads x 16B.
  {
    int c = tid >> 7;                      // eps index
    int o = (tid & 127) * 16;              // byte offset within 2KB chunk
    unsigned long long gsrc = (unsigned long long)(const char*)(ws + OFF_EB)
                            + (unsigned long long)c * (64 * TDIM * 2)
                            + (unsigned long long)m0 * (TDIM * 2) + o;
    unsigned ldsoff = (unsigned)(size_t)(&sE[0]) + (unsigned)(tid * 16);
    asm volatile("global_load_async_to_lds_b128 %0, %1, off"
                 :: "v"(ldsoff), "v"(gsrc) : "memory");
  }

  for (int i = tid; i < MROWS * TDIM; i += 512) {
    sY[i] = theta[m0 * TDIM + i];
    sKc[i] = 0.0f;
  }
  if (tid < MROWS) { sYlp[tid] = 0.0f; sKclp[tid] = 0.0f; }
  asm volatile("s_wait_asynccnt 0" ::: "memory");
  __syncthreads();

  const float dt = 1.0f / (float)NSTEP;

  for (int step = 0; step < NSTEP; ++step) {
    float tn = -1.0f + dt * (float)step;
    for (int s = 0; s < 4; ++s) {
      float cs = (s == 0) ? 0.0f : ((s == 3) ? dt : 0.5f * dt);
      float ts = tn + cs;
      float w  = (s == 1 || s == 2) ? 2.0f : 1.0f;

      // ---- build stage input X = [z_stage, ctx, t, pad] in sA (bf16) ----
      for (int i = tid; i < MROWS * INPAD; i += 512) {
        int m = i / INPAD, k = i - m * INPAD;
        float v;
        if (k < TDIM)      v = sY[m * TDIM + k] + cs * sKc[m * TDIM + k];
        else if (k < 192)  v = ctx[(m0 + m) * CDIM + (k - TDIM)];
        else if (k == 192) v = ts;
        else               v = 0.0f;
        sA[i] = f2bf(v);
      }
      if (tid < MROWS) sTR[tid] = 0.0f;
      __syncthreads();

      // ---- L1: H1 = tanh(X @ W1 + b1) ----
      for (int i = 0; i < 2; ++i) {
        int n0 = (wid * 2 + i) * 16;
        f32x8 acc = {};
#pragma unroll
        for (int k0 = 0; k0 < INPAD; k0 += 32)
          acc = wmma_bf16(load_a16(sA, INPAD, k0, lane),
                          load_b16(w1t, INPAD, n0, k0, lane), acc);
        int n = n0 + nlo;
        float bias = b1[n];
#pragma unroll
        for (int r = 0; r < 8; ++r)
          sH1[(mb + r) * HID + n] = f2bf(htanh(acc[r] + bias));
      }
      __syncthreads();

      // ---- L2: H2 = tanh(H1 @ W2 + b2) ----
      for (int i = 0; i < 2; ++i) {
        int n0 = (wid * 2 + i) * 16;
        f32x8 acc = {};
#pragma unroll
        for (int k0 = 0; k0 < HID; k0 += 32)
          acc = wmma_bf16(load_a16(sH1, HID, k0, lane),
                          load_b16(w2t, HID, n0, k0, lane), acc);
        int n = n0 + nlo;
        float bias = b2[n];
#pragma unroll
        for (int r = 0; r < 8; ++r)
          sH2[(mb + r) * HID + n] = f2bf(htanh(acc[r] + bias));
      }
      __syncthreads();

      // ---- L3: vf = H2 @ W3 + b3  (N=64 -> waves 0..3) ----
      if (wid < 4) {
        int n0 = wid * 16;
        f32x8 acc = {};
#pragma unroll
        for (int k0 = 0; k0 < HID; k0 += 32)
          acc = wmma_bf16(load_a16(sH2, HID, k0, lane),
                          load_b16(w3t, HID, n0, k0, lane), acc);
        int n = n0 + nlo;
        float bias = b3[n];
#pragma unroll
        for (int r = 0; r < 8; ++r)
          sF[(mb + r) * TDIM + n] = acc[r] + bias;
      }
      __syncthreads();

      // ---- VJP trace: for each eps, e^T J_vf e ----
      for (int e = 0; e < NEPS; ++e) {
        const unsigned short* ep = sE + e * (MROWS * TDIM);
        // bwd1: G2' = (e @ W3^T) * (1 - H2^2)  -> sA
        for (int i = 0; i < 2; ++i) {
          int n0 = (wid * 2 + i) * 16;
          f32x8 acc = {};
#pragma unroll
          for (int k0 = 0; k0 < TDIM; k0 += 32)
            acc = wmma_bf16(load_a16(ep, TDIM, k0, lane),
                            load_b16(w3b, TDIM, n0, k0, lane), acc);
          int n = n0 + nlo;
#pragma unroll
          for (int r = 0; r < 8; ++r) {
            float h2 = bf2f(sH2[(mb + r) * HID + n]);
            sA[(mb + r) * HID + n] = f2bf(acc[r] * (1.0f - h2 * h2));
          }
        }
        __syncthreads();
        // bwd2: G1' = (G2' @ W2^T) * (1 - H1^2)  -> sB
        for (int i = 0; i < 2; ++i) {
          int n0 = (wid * 2 + i) * 16;
          f32x8 acc = {};
#pragma unroll
          for (int k0 = 0; k0 < HID; k0 += 32)
            acc = wmma_bf16(load_a16(sA, HID, k0, lane),
                            load_b16(w2b, HID, n0, k0, lane), acc);
          int n = n0 + nlo;
#pragma unroll
          for (int r = 0; r < 8; ++r) {
            float h1 = bf2f(sH1[(mb + r) * HID + n]);
            sB[(mb + r) * HID + n] = f2bf(acc[r] * (1.0f - h1 * h1));
          }
        }
        __syncthreads();
        // bwd3: uz = G1' @ W1z^T ; tr += dot(uz, e)   (N=64 -> waves 0..3)
        if (wid < 4) {
          int n0 = wid * 16;
          f32x8 acc = {};
#pragma unroll
          for (int k0 = 0; k0 < HID; k0 += 32)
            acc = wmma_bf16(load_a16(sB, HID, k0, lane),
                            load_b16(w1zb, HID, n0, k0, lane), acc);
          int n = n0 + nlo;
#pragma unroll
          for (int r = 0; r < 8; ++r) {
            float ev = bf2f(ep[(mb + r) * TDIM + n]);
            atomicAdd(&sTR[mb + r], acc[r] * ev);
          }
        }
        __syncthreads();
      }

      // ---- RK4 stage update: k = (-vf, +mean_e e^T J e) ----
      for (int i = tid; i < MROWS * TDIM; i += 512) {
        float kz = -sF[i];
        sKc[i] = kz;
        sKa[i] = (s == 0) ? kz : sKa[i] + w * kz;
        if (s == 3) sY[i] += (dt / 6.0f) * sKa[i];
      }
      if (tid < MROWS) {
        float kl = sTR[tid] * 0.25f;
        sKclp[tid] = kl;
        sKalp[tid] = (s == 0) ? kl : sKalp[tid] + w * kl;
        if (s == 3) sYlp[tid] += (dt / 6.0f) * sKalp[tid];
      }
      __syncthreads();
    }
  }

  // ---- finalize: log N(z_final) - delta_log_p ----
  if (tid < MROWS) {
    float ss = 0.0f;
    for (int d = 0; d < TDIM; ++d) { float z = sY[tid * TDIM + d]; ss += z * z; }
    const float LOG2PI = 1.8378770664093453f;
    out[m0 + tid] = -0.5f * ss - 32.0f * LOG2PI - sYlp[tid];
  }
}

extern "C" void kernel_launch(void* const* d_in, const int* in_sizes, int n_in,
                              void* d_out, int out_size, void* d_ws, size_t ws_size,
                              hipStream_t stream) {
  (void)in_sizes; (void)n_in; (void)out_size; (void)ws_size;
  const float* theta = (const float*)d_in[0];
  const float* ctx   = (const float*)d_in[1];
  const float* eps   = (const float*)d_in[2];
  const float* W1    = (const float*)d_in[3];
  const float* b1    = (const float*)d_in[4];
  const float* W2    = (const float*)d_in[5];
  const float* b2    = (const float*)d_in[6];
  const float* W3    = (const float*)d_in[7];
  const float* b3    = (const float*)d_in[8];
  unsigned short* ws = (unsigned short*)d_ws;
  float* out = (float*)d_out;

  cnf_prep<<<1024, 256, 0, stream>>>(W1, W2, W3, eps, ws);
  cnf_solve<<<4, 512, 0, stream>>>(theta, ctx, b1, b2, b3, ws, out);
}
